// Dominant_13932873909002
// MI455X (gfx1250) — matrix operand order repaired and checked by reading, not compiled, not run on hardware
//
#include <hip/hip_runtime.h>

typedef __bf16 bf16_t;
typedef bf16_t v4bf  __attribute__((ext_vector_type(4)));
typedef bf16_t v8bf  __attribute__((ext_vector_type(8)));
typedef bf16_t v16bf __attribute__((ext_vector_type(16)));
typedef float  v8f   __attribute__((ext_vector_type(8)));

#define N_NODES 8192
#define FEATD   768
#define HIDD    512
#define D_ROWD  512
#define D_NULLD 256

// ---------------------------------------------------------------------------
// CDNA5 async global->LDS copy (ASYNCcnt-tracked, bypasses VGPRs).
// Generic pointers to __shared__ truncate to the LDS byte offset (ISA: the
// LDS aperture maps addr[31:0] directly to the LDS address).
// ---------------------------------------------------------------------------
__device__ __forceinline__ void async_ld16(void* lds, const void* gaddr)
{
    asm volatile("global_load_async_to_lds_b128 %0, %1, off"
                 :: "v"((unsigned)(uintptr_t)lds), "v"(gaddr)
                 : "memory");
}
__device__ __forceinline__ void wait_async0()
{
    asm volatile("s_wait_asynccnt 0" ::: "memory");
}

// ---------------------------------------------------------------------------
// Elementwise conversion kernels
// ---------------------------------------------------------------------------
__global__ void cvt_bf16_k(const float* __restrict__ s, bf16_t* __restrict__ d, int n)
{
    int i = (blockIdx.x * blockDim.x + threadIdx.x) * 4;
    if (i < n) {
        float4 f = *(const float4*)(s + i);
        v4bf o;
        o[0] = (bf16_t)f.x; o[1] = (bf16_t)f.y; o[2] = (bf16_t)f.z; o[3] = (bf16_t)f.w;
        *(v4bf*)(d + i) = o;
    }
}

__global__ void center_bf16_k(const float* __restrict__ te, const float* __restrict__ mu,
                              bf16_t* __restrict__ d, int n)
{
    int i = (blockIdx.x * blockDim.x + threadIdx.x) * 4;
    if (i < n) {
        float4 f = *(const float4*)(te + i);
        int c = i % FEATD;                      // 768 % 4 == 0 -> no row wrap
        float4 m = *(const float4*)(mu + c);
        v4bf o;
        o[0] = (bf16_t)(f.x - m.x); o[1] = (bf16_t)(f.y - m.y);
        o[2] = (bf16_t)(f.z - m.z); o[3] = (bf16_t)(f.w - m.w);
        *(v4bf*)(d + i) = o;
    }
}

// dst (C x R) = transpose of src (R x C), fp32 -> bf16
__global__ void tr_bf16_k(const float* __restrict__ s, bf16_t* __restrict__ d, int R, int C)
{
    int g = blockIdx.x * blockDim.x + threadIdx.x;
    if (g < R * C) {
        int r = g / C, c = g - r * C;
        d[(size_t)c * R + r] = (bf16_t)s[g];
    }
}

__global__ void colmean_k(const float* __restrict__ te, float* __restrict__ mu)
{
    int c = blockIdx.x * blockDim.x + threadIdx.x;
    if (c < FEATD) {
        float s = 0.0f;
        for (int r = 0; r < N_NODES; ++r) s += te[(size_t)r * FEATD + c];
        mu[c] = s * (1.0f / (float)N_NODES);
    }
}

// ---------------------------------------------------------------------------
// Covariance (fp32 for eigen-precision): C[i][j] = sum_n Xc[n][i]*Xc[n][j]/(N-1)
// ---------------------------------------------------------------------------
__global__ __launch_bounds__(256) void cov_k(const float* __restrict__ te,
                                             const float* __restrict__ mu,
                                             float* __restrict__ C)
{
    __shared__ float Ab[32][65];
    __shared__ float Bb[32][65];
    const int i0 = blockIdx.y * 64, j0 = blockIdx.x * 64;
    const int tid = threadIdx.x;
    const int lr = tid >> 3;
    const int lc = (tid & 7) * 8;
    const int ty = tid >> 4, tx = tid & 15;

    float acc[4][4];
#pragma unroll
    for (int u = 0; u < 4; ++u)
#pragma unroll
        for (int v = 0; v < 4; ++v) acc[u][v] = 0.0f;

    for (int n0 = 0; n0 < N_NODES; n0 += 32) {
#pragma unroll
        for (int e = 0; e < 8; ++e) {
            Ab[lr][lc + e] = te[(size_t)(n0 + lr) * FEATD + i0 + lc + e] - mu[i0 + lc + e];
            Bb[lr][lc + e] = te[(size_t)(n0 + lr) * FEATD + j0 + lc + e] - mu[j0 + lc + e];
        }
        __syncthreads();
#pragma unroll 8
        for (int kk = 0; kk < 32; ++kk) {
            float a[4], b[4];
#pragma unroll
            for (int u = 0; u < 4; ++u) a[u] = Ab[kk][ty * 4 + u];
#pragma unroll
            for (int v = 0; v < 4; ++v) b[v] = Bb[kk][tx * 4 + v];
#pragma unroll
            for (int u = 0; u < 4; ++u)
#pragma unroll
                for (int v = 0; v < 4; ++v) acc[u][v] += a[u] * b[v];
        }
        __syncthreads();
    }
    const float inv = 1.0f / (float)(N_NODES - 1);
#pragma unroll
    for (int u = 0; u < 4; ++u)
#pragma unroll
        for (int v = 0; v < 4; ++v)
            C[(size_t)(i0 + ty * 4 + u) * FEATD + j0 + tx * 4 + v] = acc[u][v] * inv;
}

// ---------------------------------------------------------------------------
// Parallel cyclic Jacobi eigensolver (cooperative kernel, grid barrier)
// ---------------------------------------------------------------------------
#define JB 96
#define JT 256
#define JSWEEPS 6

__global__ void init_eig_k(float* __restrict__ U, unsigned* __restrict__ bar)
{
    int g = blockIdx.x * blockDim.x + threadIdx.x;
    if (g < FEATD * FEATD) {
        int r = g / FEATD, c = g - r * FEATD;
        U[g] = (r == c) ? 1.0f : 0.0f;
    }
    if (g < 2) bar[g] = 0u;
}

__device__ __forceinline__ void jpair(int r, int i, int* p, int* q)
{
    const int m = FEATD - 1;                    // 767 (odd -> pairs always distinct)
    int a, b;
    if (i == 0) { a = m; b = r % m; }
    else        { a = (r + i) % m; b = (r - i + m) % m; }
    if (a < b) { *p = a; *q = b; } else { *p = b; *q = a; }
}

__device__ __forceinline__ void gbar(volatile unsigned* bar)
{
    __syncthreads();
    if (threadIdx.x == 0) {
        __threadfence();
        unsigned gen = bar[1];
        if (atomicAdd((unsigned*)&bar[0], 1u) == (unsigned)(JB - 1)) {
            bar[0] = 0u;
            __threadfence();
            bar[1] = gen + 1u;
        } else {
            while (bar[1] == gen) { __builtin_amdgcn_s_sleep(2); }
        }
        __threadfence();
    }
    __syncthreads();
}

__global__ __launch_bounds__(JT) void jacobi_k(float* __restrict__ A, float* __restrict__ U,
                                               float* __restrict__ cA, float* __restrict__ sA,
                                               unsigned* bar)
{
    const int gid = blockIdx.x * JT + threadIdx.x;
    const int nt  = JB * JT;                    // 24576; 384*768 = 12*nt exactly
    for (int sweep = 0; sweep < JSWEEPS; ++sweep) {
        for (int r = 0; r < FEATD - 1; ++r) {
            // phase A: rotation angles for 384 disjoint pairs
            for (int i = gid; i < FEATD / 2; i += nt) {
                int p, q; jpair(r, i, &p, &q);
                float app = A[(size_t)p * FEATD + p];
                float aqq = A[(size_t)q * FEATD + q];
                float apq = A[(size_t)p * FEATD + q];
                float c = 1.0f, s = 0.0f;
                if (fabsf(apq) > 1e-30f) {
                    float tau = (aqq - app) / (2.0f * apq);
                    float t = copysignf(1.0f, tau) / (fabsf(tau) + sqrtf(1.0f + tau * tau));
                    c = 1.0f / sqrtf(1.0f + t * t);
                    s = t * c;
                }
                cA[i] = c; sA[i] = s;
            }
            gbar(bar);
            // phase B: A <- J^T A   (row rotations)
            for (int t = gid; t < (FEATD / 2) * FEATD; t += nt) {
                int i = t / FEATD, j = t - i * FEATD;
                int p, q; jpair(r, i, &p, &q);
                float c = cA[i], s = sA[i];
                float ap = A[(size_t)p * FEATD + j], aq = A[(size_t)q * FEATD + j];
                A[(size_t)p * FEATD + j] = c * ap - s * aq;
                A[(size_t)q * FEATD + j] = s * ap + c * aq;
            }
            gbar(bar);
            // phase C: A <- A J, U <- U J  (column rotations)
            for (int t = gid; t < (FEATD / 2) * FEATD; t += nt) {
                int i = t / FEATD, j = t - i * FEATD;
                int p, q; jpair(r, i, &p, &q);
                float c = cA[i], s = sA[i];
                float ap = A[(size_t)j * FEATD + p], aq = A[(size_t)j * FEATD + q];
                A[(size_t)j * FEATD + p] = c * ap - s * aq;
                A[(size_t)j * FEATD + q] = s * ap + c * aq;
                float up = U[(size_t)j * FEATD + p], uq = U[(size_t)j * FEATD + q];
                U[(size_t)j * FEATD + p] = c * up - s * uq;
                U[(size_t)j * FEATD + q] = s * up + c * uq;
            }
            gbar(bar);
        }
    }
}

// one block of 768 threads: even-odd transposition sort, descending eigenvalues
__global__ __launch_bounds__(FEATD) void eig_sort_k(const float* __restrict__ A,
                                                    float* __restrict__ lam, int* __restrict__ idx)
{
    __shared__ float v[FEATD];
    __shared__ int ix[FEATD];
    const int t = threadIdx.x;
    v[t] = A[(size_t)t * FEATD + t];
    ix[t] = t;
    __syncthreads();
    for (int pass = 0; pass < FEATD; ++pass) {
        int i = 2 * t + (pass & 1);
        if (i + 1 < FEATD) {
            if (v[i] < v[i + 1]) {
                float tv = v[i]; v[i] = v[i + 1]; v[i + 1] = tv;
                int ti = ix[i]; ix[i] = ix[i + 1]; ix[i + 1] = ti;
            }
        }
        __syncthreads();
    }
    lam[t] = v[t];
    idx[t] = ix[t];
}

// P_rowT[n][k] = U[k][idx[n]] / sqrt(lam[n] + eps)  (bf16, 512x768)
__global__ void build_prowT_k(const float* __restrict__ U, const float* __restrict__ lam,
                              const int* __restrict__ idx, bf16_t* __restrict__ P)
{
    int g = blockIdx.x * blockDim.x + threadIdx.x;
    if (g < D_ROWD * FEATD) {
        int n = g / FEATD, k = g - n * FEATD;
        float scl = rsqrtf(lam[n] + 1e-12f);
        P[g] = (bf16_t)(U[(size_t)k * FEATD + idx[n]] * scl);
    }
}

// ---------------------------------------------------------------------------
// WMMA bf16 NT GEMM: C(MxN) = act(A(MxK) * B(NxK)^T + bias[n])
// 128x128 tile per 256-thread block (8 waves, 2x4 wave grid, 4x2 WMMA tiles).
// Double-buffered LDS tiles filled with global_load_async_to_lds_b128:
// issue tile k+1, run 16 WMMAs/wave on tile k, s_wait_asynccnt 0, barrier.
// ---------------------------------------------------------------------------
#define GT_M 128
#define GT_N 128
#define GT_K 64
#define LPAD 8

template <bool RELU, bool OUTBF16>
__global__ __launch_bounds__(256) void gemm_nt_k(const bf16_t* __restrict__ A, int lda,
                                                 const bf16_t* __restrict__ B, int ldb,
                                                 void* __restrict__ Cout, int ldo,
                                                 const float* __restrict__ bias, int K)
{
    __shared__ bf16_t As[2][GT_M][GT_K + LPAD];
    __shared__ bf16_t Bs[2][GT_N][GT_K + LPAD];

    const int tid  = threadIdx.x;
    const int bm   = blockIdx.y * GT_M;
    const int bn   = blockIdx.x * GT_N;
    const int wave = tid >> 5;
    const int lane = tid & 31;
    const int wr   = wave >> 2;     // 0..1 : 64-row half of the M tile
    const int wc   = wave & 3;      // 0..3 : 32-col quarter of the N tile
    const int lh   = lane >> 4;     // K-phase half per ISA A/B layouts
    const int l16  = lane & 15;

    v8f acc[4][2];
#pragma unroll
    for (int i = 0; i < 4; ++i)
#pragma unroll
        for (int j = 0; j < 2; ++j)
#pragma unroll
            for (int e = 0; e < 8; ++e) acc[i][j][e] = 0.0f;

    const int lrow = tid >> 1;            // 0..127
    const int lcol = (tid & 1) * 32;      // 0 or 32
    const bf16_t* Ag = A + (size_t)(bm + lrow) * lda + lcol;
    const bf16_t* Bg = B + (size_t)(bn + lrow) * ldb + lcol;

    // issue one 64-deep K tile (8 async b128 copies per thread)
    auto issue_tile = [&](int k0, int buf) {
#pragma unroll
        for (int c = 0; c < 4; ++c) {
            async_ld16(&As[buf][lrow][lcol + c * 8], Ag + k0 + c * 8);
            async_ld16(&Bs[buf][lrow][lcol + c * 8], Bg + k0 + c * 8);
        }
    };

    issue_tile(0, 0);
    wait_async0();
    __syncthreads();

    const int nk = K / GT_K;
    for (int kt = 0; kt < nk; ++kt) {
        const int cur = kt & 1;
        if (kt + 2 < nk) {                // prefetch one tile beyond the async copy
            __builtin_prefetch(Ag + (kt + 2) * GT_K, 0, 3);
            __builtin_prefetch(Bg + (kt + 2) * GT_K, 0, 3);
        }
        if (kt + 1 < nk) issue_tile((kt + 1) * GT_K, cur ^ 1);

#pragma unroll
        for (int ks = 0; ks < GT_K; ks += 32) {
            // A fragment: lane<16 holds M=l16, K={0..7,16..23}; lane>=16 K={8..15,24..31}
            v16bf af[4];
#pragma unroll
            for (int i = 0; i < 4; ++i) {
                const bf16_t* pa = &As[cur][wr * 64 + i * 16 + l16][ks + lh * 8];
                v8bf lo = *(const v8bf*)pa;
                v8bf hi = *(const v8bf*)(pa + 16);
#pragma unroll
                for (int e = 0; e < 8; ++e) { af[i][e] = lo[e]; af[i][8 + e] = hi[e]; }
            }
            // B fragment: lane<16 holds N=l16, K={0..7,16..23}; lane>=16 K={8..15,24..31}
            v16bf bfr[2];
#pragma unroll
            for (int j = 0; j < 2; ++j) {
                const bf16_t* pb = &Bs[cur][wc * 32 + j * 16 + l16][ks + lh * 8];
                v8bf lo = *(const v8bf*)pb;
                v8bf hi = *(const v8bf*)(pb + 16);
#pragma unroll
                for (int e = 0; e < 8; ++e) { bfr[j][e] = lo[e]; bfr[j][8 + e] = hi[e]; }
            }
#pragma unroll
            for (int i = 0; i < 4; ++i)
#pragma unroll
                for (int j = 0; j < 2; ++j)
                    acc[i][j] = __builtin_amdgcn_wmma_f32_16x16x32_bf16(
                        false, af[i], false, bfr[j], (short)0, acc[i][j], false, false);
        }
        wait_async0();                    // next tile landed; also fences reads of cur
        __syncthreads();
    }

    // epilogue: C/D layout -> lane<16:(M=r,N=l16), lane>=16:(M=8+r,N=l16)
#pragma unroll
    for (int j = 0; j < 2; ++j) {
        const int n = bn + wc * 32 + j * 16 + l16;
        float bv = 0.0f;
        if (bias) bv = bias[n];
#pragma unroll
        for (int i = 0; i < 4; ++i) {
            const int mbase = bm + wr * 64 + i * 16 + lh * 8;
#pragma unroll
            for (int r = 0; r < 8; ++r) {
                float v = acc[i][j][r] + bv;
                if (RELU) v = fmaxf(v, 0.0f);
                if (OUTBF16) ((bf16_t*)Cout)[(size_t)(mbase + r) * ldo + n] = (bf16_t)v;
                else         ((float*)Cout)[(size_t)(mbase + r) * ldo + n] = v;
            }
        }
    }
}

// ---------------------------------------------------------------------------
// Host orchestration
// ---------------------------------------------------------------------------
extern "C" void kernel_launch(void* const* d_in, const int* in_sizes, int n_in,
                              void* d_out, int out_size, void* d_ws, size_t ws_size,
                              hipStream_t stream)
{
    (void)in_sizes; (void)n_in; (void)out_size; (void)ws_size;
    const float* x_f   = (const float*)d_in[0];
    const float* adj_f = (const float*)d_in[1];
    const float* te_f  = (const float*)d_in[2];
    const float* We1   = (const float*)d_in[3];
    const float* be1   = (const float*)d_in[4];
    const float* We2   = (const float*)d_in[5];
    const float* be2   = (const float*)d_in[6];
    const float* Wa1   = (const float*)d_in[7];
    const float* ba1   = (const float*)d_in[8];
    const float* Wa2   = (const float*)d_in[9];
    const float* ba2   = (const float*)d_in[10];
    const float* Ws1   = (const float*)d_in[11];
    const float* bs1   = (const float*)d_in[12];
    const float* Wproj = (const float*)d_in[13];
    float* out = (float*)d_out;

    char* w = (char*)d_ws;
    auto carve = [&](size_t bytes) -> char* {
        char* p = w; w += (bytes + 255) & ~(size_t)255; return p;
    };

    bf16_t* adjB   = (bf16_t*)carve((size_t)N_NODES * N_NODES * 2);      // 128 MB, L2-resident
    bf16_t* XcB    = (bf16_t*)carve((size_t)N_NODES * FEATD * 2);
    bf16_t* xB     = (bf16_t*)carve((size_t)N_NODES * FEATD * 2);
    bf16_t* Zb     = (bf16_t*)carve((size_t)N_NODES * FEATD * 2);        // [X_row | H_null]
    bf16_t* Tt     = (bf16_t*)carve((size_t)FEATD * N_NODES * 2);        // reused (h W)^T buffer
    bf16_t* hB     = (bf16_t*)carve((size_t)N_NODES * HIDD * 2);
    bf16_t* h2B    = (bf16_t*)carve((size_t)N_NODES * HIDD * 2);
    bf16_t* xhB    = (bf16_t*)carve((size_t)N_NODES * HIDD * 2);
    bf16_t* sB     = (bf16_t*)carve((size_t)N_NODES * HIDD * 2);
    float*  Aw     = (float*)carve((size_t)FEATD * FEATD * 4);           // Jacobi work matrix
    float*  Uw     = (float*)carve((size_t)FEATD * FEATD * 4);           // eigenvectors
    float*  mu     = (float*)carve(FEATD * 4);
    float*  lam    = (float*)carve(FEATD * 4);
    int*    idxv   = (int*)carve(FEATD * 4);
    float*  csA    = (float*)carve((FEATD / 2) * 4);
    float*  snA    = (float*)carve((FEATD / 2) * 4);
    unsigned* bar  = (unsigned*)carve(256);
    bf16_t* ProwT  = (bf16_t*)carve((size_t)D_ROWD * FEATD * 2);
    bf16_t* We1T   = (bf16_t*)carve((size_t)HIDD * FEATD * 2);
    bf16_t* We2T   = (bf16_t*)carve((size_t)HIDD * HIDD * 2);
    bf16_t* Wa1T   = (bf16_t*)carve((size_t)HIDD * HIDD * 2);
    bf16_t* Ws1T   = (bf16_t*)carve((size_t)HIDD * HIDD * 2);
    bf16_t* Wa2T   = (bf16_t*)carve((size_t)FEATD * HIDD * 2);
    bf16_t* WprojB = (bf16_t*)carve((size_t)D_NULLD * FEATD * 2);

    const dim3 blk(256);

    // --- precision prep ---
    cvt_bf16_k<<<65536, blk, 0, stream>>>(adj_f, adjB, N_NODES * N_NODES);
    cvt_bf16_k<<<6144, blk, 0, stream>>>(x_f, xB, N_NODES * FEATD);
    cvt_bf16_k<<<192, blk, 0, stream>>>(Wproj, WprojB, D_NULLD * FEATD);
    colmean_k<<<3, blk, 0, stream>>>(te_f, mu);
    center_bf16_k<<<6144, blk, 0, stream>>>(te_f, mu, XcB, N_NODES * FEATD);
    tr_bf16_k<<<1536, blk, 0, stream>>>(We1, We1T, FEATD, HIDD);
    tr_bf16_k<<<1024, blk, 0, stream>>>(We2, We2T, HIDD, HIDD);
    tr_bf16_k<<<1024, blk, 0, stream>>>(Wa1, Wa1T, HIDD, HIDD);
    tr_bf16_k<<<1024, blk, 0, stream>>>(Ws1, Ws1T, HIDD, HIDD);
    tr_bf16_k<<<1536, blk, 0, stream>>>(Wa2, Wa2T, HIDD, FEATD);

    // --- covariance + eigendecomposition (fp32) ---
    cov_k<<<dim3(12, 12), blk, 0, stream>>>(te_f, mu, Aw);
    init_eig_k<<<2304, blk, 0, stream>>>(Uw, bar);
    jacobi_k<<<JB, JT, 0, stream>>>(Aw, Uw, csA, snA, bar);
    eig_sort_k<<<1, FEATD, 0, stream>>>(Aw, lam, idxv);
    build_prowT_k<<<1536, blk, 0, stream>>>(Uw, lam, idxv, ProwT);

    // --- GEMM pipeline (all NT: C = A(MxK) * B(NxK)^T) ---
    // Z[:, :512]  = Xc @ P_row
    gemm_nt_k<false, true><<<dim3(4, 64), blk, 0, stream>>>(XcB, FEATD, ProwT, FEATD,
                                                            (void*)Zb, FEATD, nullptr, FEATD);
    // Z[:, 512:]  = x @ Wproj^T
    gemm_nt_k<false, true><<<dim3(2, 64), blk, 0, stream>>>(xB, FEATD, WprojB, FEATD,
                                                            (void*)(Zb + D_ROWD), FEATD, nullptr, FEATD);
    // T1^T = We1^T @ Z^T  (512 x 8192)
    gemm_nt_k<false, true><<<dim3(64, 4), blk, 0, stream>>>(We1T, FEATD, Zb, FEATD,
                                                            (void*)Tt, N_NODES, nullptr, FEATD);
    // h = relu(adj @ T1 + be1)
    gemm_nt_k<true, true><<<dim3(4, 64), blk, 0, stream>>>(adjB, N_NODES, Tt, N_NODES,
                                                           (void*)hB, HIDD, be1, N_NODES);
    // T2^T = We2^T @ h^T
    gemm_nt_k<false, true><<<dim3(64, 4), blk, 0, stream>>>(We2T, HIDD, hB, HIDD,
                                                            (void*)Tt, N_NODES, nullptr, HIDD);
    // h2 = relu(adj @ T2 + be2)
    gemm_nt_k<true, true><<<dim3(4, 64), blk, 0, stream>>>(adjB, N_NODES, Tt, N_NODES,
                                                           (void*)h2B, HIDD, be2, N_NODES);
    // T3^T = Wa1^T @ h2^T
    gemm_nt_k<false, true><<<dim3(64, 4), blk, 0, stream>>>(Wa1T, HIDD, h2B, HIDD,
                                                            (void*)Tt, N_NODES, nullptr, HIDD);
    // xh = relu(adj @ T3 + ba1)
    gemm_nt_k<true, true><<<dim3(4, 64), blk, 0, stream>>>(adjB, N_NODES, Tt, N_NODES,
                                                           (void*)xhB, HIDD, ba1, N_NODES);
    // T4^T = Wa2^T @ xh^T  (768 x 8192)
    gemm_nt_k<false, true><<<dim3(64, 6), blk, 0, stream>>>(Wa2T, HIDD, xhB, HIDD,
                                                            (void*)Tt, N_NODES, nullptr, HIDD);
    // x_hat = relu(adj @ T4 + ba2)  -> fp32 output tail
    gemm_nt_k<true, false><<<dim3(6, 64), blk, 0, stream>>>(adjB, N_NODES, Tt, N_NODES,
                                                            (void*)(out + (size_t)N_NODES * N_NODES),
                                                            FEATD, ba2, N_NODES);
    // T5^T = Ws1^T @ h2^T
    gemm_nt_k<false, true><<<dim3(64, 4), blk, 0, stream>>>(Ws1T, HIDD, h2B, HIDD,
                                                            (void*)Tt, N_NODES, nullptr, HIDD);
    // s = relu(adj @ T5 + bs1)
    gemm_nt_k<true, true><<<dim3(4, 64), blk, 0, stream>>>(adjB, N_NODES, Tt, N_NODES,
                                                           (void*)sB, HIDD, bs1, N_NODES);
    // struct = s @ s^T  -> fp32 output head (8192 x 8192)
    gemm_nt_k<false, false><<<dim3(64, 64), blk, 0, stream>>>(sB, HIDD, sB, HIDD,
                                                              (void*)out, N_NODES, nullptr, HIDD);
}
